// WindowAttention_37546604101788
// MI455X (gfx1250) — matrix operand order repaired and checked by reading, not compiled, not run on hardware
//
#include <hip/hip_runtime.h>

typedef __bf16 bf16;
typedef __attribute__((ext_vector_type(16))) __bf16 bf16x16;
typedef __attribute__((ext_vector_type(8)))  float  f32x8;
typedef __attribute__((ext_vector_type(4)))  float  f32x4;

#define DIMC   256
#define HEADS  8
#define HD     32
#define NTOK   49
#define NPAD   64
#define BWIN   4096
#define NWM    64
#define MROWS  (BWIN * NTOK)      // 200704 = 64 * 3136
#define MTILES (MROWS / 64)       // 3136
#define QSCALE 0.1767766952966369f
#define PERQ   ((size_t)BWIN * HEADS * NTOK * HD)   // 51,380,224 elements

// ---------------- async global->LDS (ASYNCcnt path), guarded per-toolchain ----------------
#if __has_builtin(__builtin_amdgcn_global_load_async_to_lds_b128) && \
    __has_builtin(__builtin_amdgcn_s_wait_asynccnt)
#define USE_ASYNC 1
typedef int i32x4v __attribute__((vector_size(16)));
__device__ __forceinline__ void async_cp16(void* lds, const void* g) {
    __builtin_amdgcn_global_load_async_to_lds_b128(
        (__attribute__((address_space(1))) i32x4v*)g,
        (__attribute__((address_space(3))) i32x4v*)lds, 0, 0);
}
__device__ __forceinline__ void async_wait_all() { __builtin_amdgcn_s_wait_asynccnt(0); }
#else
#define USE_ASYNC 0
#endif

// ---------------- WMMA fragment loaders (per CDNA5 ISA VGPR layouts) ----------------

// A-matrix 16x32 bf16 fragment from row-major storage (row stride in elements).
// lanes 0-15: row = lane, K = {0..7, 16..23}; lanes 16-31: row = lane-16, K = {8..15, 24..31}
__device__ __forceinline__ bf16x16 load_a_frag(const bf16* base, int ld, int lane) {
    int row = lane & 15;
    int k0  = (lane >> 4) * 8;
    const bf16* p = base + row * ld + k0;
    bf16x16 f;
#pragma unroll
    for (int i = 0; i < 8; ++i) f[i] = p[i];
#pragma unroll
    for (int i = 0; i < 8; ++i) f[8 + i] = p[16 + i];
    return f;
}

// B-matrix 32x16 bf16 fragment from TRANSPOSED storage baseT[col][k] (row stride ld).
// lanes 0-15: col = lane, K = {0..7, 16..23}; lanes 16-31: col = lane-16, K = {8..15, 24..31}
__device__ __forceinline__ bf16x16 load_bt_frag(const bf16* baseT, int ld, int lane) {
    int col = lane & 15;
    int k0  = (lane >> 4) * 8;
    const bf16* p = baseT + col * ld + k0;
    bf16x16 f;
#pragma unroll
    for (int i = 0; i < 16; ++i) f[i] = p[(i < 8) ? i : (i + 8)];
    return f;
}

// ---------------- Kernel 1: QKV projection  [M,256] x [256,768] -> q,k,v ----------------
// 256 threads (8 wave32). Block computes 64 rows x all 768 cols, so x is read ONCE.
// Wave w: row strip (w&3)*16, column half (w>>2)*384 -> 24 accumulator tiles.

__global__ __launch_bounds__(256)
void qkv_gemm_kernel(const float* __restrict__ x, const float* __restrict__ wqkv,
                     bf16* __restrict__ qkv_ws)
{
    __shared__ bf16 As[64][40];    // 64 rows x 32 K (80B rows: b128-friendly)
    __shared__ bf16 Bt[768][36];   // 768 cols x 32 K transposed (72B rows: 8B aligned)

    const int row0 = blockIdx.x * 64;
    const int t    = threadIdx.x;
    const int w    = t >> 5;
    const int lane = t & 31;
    const int mt   = (w & 3) * 16;        // row strip
    const int ct0  = (w >> 2) * 24;       // first column tile (of 16 cols)

    f32x8 acc[24];
#pragma unroll
    for (int i = 0; i < 24; ++i) acc[i] = {};

    for (int ks = 0; ks < 8; ++ks) {
        const int kk0 = ks * 32;
        // stage A: 64x32 f32 -> bf16 (float4 vector loads); 512 float4 / 256 threads
        for (int e = t; e < 512; e += 256) {
            int m = e >> 3, c4 = (e & 7) * 4;
            f32x4 xv = *(const f32x4*)(x + (size_t)(row0 + m) * DIMC + kk0 + c4);
            As[m][c4 + 0] = (bf16)xv.x;
            As[m][c4 + 1] = (bf16)xv.y;
            As[m][c4 + 2] = (bf16)xv.z;
            As[m][c4 + 3] = (bf16)xv.w;
        }
        // stage Bt: Bt[c][kk] = W[kk0+kk][c], all 768 cols (W stays L2-resident)
#pragma unroll 2
        for (int kk = 0; kk < 32; ++kk) {
            const float* wr = wqkv + (size_t)(kk0 + kk) * 768;
            Bt[t][kk]       = (bf16)wr[t];
            Bt[t + 256][kk] = (bf16)wr[t + 256];
            Bt[t + 512][kk] = (bf16)wr[t + 512];
        }
        if (ks < 7)
            __builtin_prefetch(x + (size_t)(row0 + (t & 63)) * DIMC + kk0 + 32, 0, 1);
        __syncthreads();

        bf16x16 af = load_a_frag(&As[mt][0], 40, lane);
#pragma unroll
        for (int nt = 0; nt < 24; ++nt) {
            bf16x16 bfp = load_bt_frag(&Bt[(ct0 + nt) * 16][0], 36, lane);
            acc[nt] = __builtin_amdgcn_wmma_f32_16x16x32_bf16(
                false, af, false, bfp, (short)0, acc[nt], false, false);
        }
        __syncthreads();
    }

    // epilogue: scatter into q/k/v [b][h][n][d] bf16 (q pre-scaled)
#pragma unroll
    for (int nt = 0; nt < 24; ++nt) {
        int cg    = (ct0 + nt) * 16 + (lane & 15);   // 0..767
        int which = cg >> 8;                         // 0=q 1=k 2=v
        int c2    = cg & 255;
        int h = c2 >> 5, d = c2 & 31;
        float scl = (which == 0) ? QSCALE : 1.0f;
        bf16* dst = qkv_ws + (size_t)which * PERQ;
#pragma unroll
        for (int r = 0; r < 8; ++r) {
            int gRow = row0 + mt + r + 8 * (lane >> 4);
            int b = gRow / NTOK, n = gRow % NTOK;
            dst[(((size_t)b * HEADS + h) * NTOK + n) * HD + d] = (bf16)(acc[nt][r] * scl);
        }
    }
}

// ---------------- Kernel 2: per-(window,head) attention ----------------

__global__ __launch_bounds__(128)
void attn_kernel(const bf16* __restrict__ q_ws, const bf16* __restrict__ k_ws,
                 const bf16* __restrict__ v_ws, const float* __restrict__ mask,
                 const float* __restrict__ bias_table, const int* __restrict__ rel_idx,
                 bf16* __restrict__ ctx)
{
    __shared__ __align__(16) bf16  qs[NPAD][HD];
    __shared__ __align__(16) bf16  ksm[NPAD][HD];
    __shared__ bf16  vt[HD][NPAD];        // V transposed: vt[d][token]
    __shared__ float S[NPAD][NPAD + 1];
    __shared__ bf16  P[NPAD][NPAD];

    const int bh = blockIdx.x;
    const int b  = bh >> 3;
    const int h  = bh & 7;
    const int t  = threadIdx.x;
    const int w  = t >> 5;
    const int lane = t & 31;

    const bf16* qg = q_ws + (size_t)bh * NTOK * HD;
    const bf16* kg = k_ws + (size_t)bh * NTOK * HD;
    const bf16* vg = v_ws + (size_t)bh * NTOK * HD;
    const float* mrow = mask + (size_t)(b & (NWM - 1)) * NTOK * NTOK;

#if USE_ASYNC
    // q,k rows 0..48 are a contiguous 3136-byte block: 196 x 16B async chunks each
    {
        char* qsb = (char*)&qs[0][0];
        char* ksb = (char*)&ksm[0][0];
        const char* qgb = (const char*)qg;
        const char* kgb = (const char*)kg;
        for (int e = t; e < 196; e += 128) {
            async_cp16(qsb + e * 16, qgb + e * 16);
            async_cp16(ksb + e * 16, kgb + e * 16);
        }
        // zero the pad rows 49..63 (bytes 3136..4095 -> 240 dwords)
        unsigned int* qz = (unsigned int*)(qsb + NTOK * HD * 2);
        unsigned int* kz = (unsigned int*)(ksb + NTOK * HD * 2);
        for (int e = t; e < 240; e += 128) { qz[e] = 0u; kz[e] = 0u; }
    }
#else
    for (int e = t; e < NPAD * HD; e += 128) {
        int m = e >> 5, d = e & 31;
        bf16 zq = (bf16)0.0f, zk = (bf16)0.0f;
        if (m < NTOK) { zq = qg[m * HD + d]; zk = kg[m * HD + d]; }
        qs[m][d] = zq;
        ksm[m][d] = zk;
    }
#endif
    // stage V transposed (zero-pad token cols)
    for (int e = t; e < HD * NPAD; e += 128) {
        int d = e >> 6, m = e & 63;
        vt[d][m] = (m < NTOK) ? vg[m * HD + d] : (bf16)0.0f;
    }
#if USE_ASYNC
    async_wait_all();
#endif
    __syncthreads();

    // S = q k^T  (wave w -> row strip w*16, 4 column tiles)
    {
        bf16x16 af = load_a_frag(&qs[w * 16][0], HD, lane);
        f32x8 sacc[4];
#pragma unroll
        for (int nt = 0; nt < 4; ++nt) {
            f32x8 z = {};
            bf16x16 bfp = load_bt_frag(&ksm[nt * 16][0], HD, lane);
            sacc[nt] = __builtin_amdgcn_wmma_f32_16x16x32_bf16(
                false, af, false, bfp, (short)0, z, false, false);
        }
#pragma unroll
        for (int nt = 0; nt < 4; ++nt) {
#pragma unroll
            for (int r = 0; r < 8; ++r) {
                int row = w * 16 + r + 8 * (lane >> 4);
                int col = nt * 16 + (lane & 15);
                float val = -1e30f;
                if (row < NTOK && col < NTOK) {
                    int ri = rel_idx[row * NTOK + col];
                    val = sacc[nt][r] + bias_table[ri * HEADS + h] + mrow[row * NTOK + col];
                }
                S[row][col] = val;
            }
        }
    }
    __syncthreads();

    // row softmax -> P (bf16)
    if (t < NPAD) {
        float mx = -3.0e38f;
#pragma unroll 8
        for (int m = 0; m < NPAD; ++m) mx = fmaxf(mx, S[t][m]);
        float sum = 0.0f;
#pragma unroll 8
        for (int m = 0; m < NPAD; ++m) { float e = __expf(S[t][m] - mx); S[t][m] = e; sum += e; }
        float inv = 1.0f / sum;
#pragma unroll 8
        for (int m = 0; m < NPAD; ++m) P[t][m] = (bf16)(S[t][m] * inv);
    }
    __syncthreads();

    // out = P @ V  (wave w -> row strip, 2 column tiles of 16, K = 64 in two 32-steps)
#pragma unroll
    for (int ct = 0; ct < 2; ++ct) {
        f32x8 oacc = {};
#pragma unroll
        for (int k2 = 0; k2 < 2; ++k2) {
            bf16x16 af  = load_a_frag(&P[w * 16][k2 * 32], NPAD, lane);
            bf16x16 bfp = load_bt_frag(&vt[ct * 16][k2 * 32], NPAD, lane);
            oacc = __builtin_amdgcn_wmma_f32_16x16x32_bf16(
                false, af, false, bfp, (short)0, oacc, false, false);
        }
#pragma unroll
        for (int r = 0; r < 8; ++r) {
            int row = w * 16 + r + 8 * (lane >> 4);
            if (row < NTOK) {
                int d = ct * 16 + (lane & 15);
                ctx[((size_t)b * NTOK + row) * DIMC + h * HD + d] = (bf16)oacc[r];
            }
        }
    }
}

// ---------------- Kernel 3: output projection  [M,256] x [256,256] + bias ----------------

__global__ __launch_bounds__(128)
void proj_gemm_kernel(const bf16* __restrict__ ctx, const float* __restrict__ wproj,
                      const float* __restrict__ bproj, float* __restrict__ out)
{
    __shared__ __align__(16) bf16 As[64][48];   // 96B rows: 16B-aligned chunks, conflict-spread
    __shared__ bf16 Bt[256][40];

    const int row0 = blockIdx.x * 64;
    const int t    = threadIdx.x;
    const int w    = t >> 5;
    const int lane = t & 31;

    f32x8 acc[16];
#pragma unroll
    for (int i = 0; i < 16; ++i) acc[i] = {};

    for (int ks = 0; ks < 8; ++ks) {
        const int kk0 = ks * 32;
#if USE_ASYNC
        // A tile is already bf16: 64 rows x 64B -> 256 x 16B async chunks
        for (int e = t; e < 256; e += 128) {
            int m = e >> 2, c = (e & 3) * 8;
            async_cp16(&As[m][c], ctx + (size_t)(row0 + m) * DIMC + kk0 + c);
        }
#else
        for (int e = t; e < 2048; e += 128) {
            int m = e >> 5, c = e & 31;
            As[m][c] = ctx[(size_t)(row0 + m) * DIMC + kk0 + c];
        }
#endif
#pragma unroll 4
        for (int kk = 0; kk < 32; ++kk) {
            const float* wr = wproj + (size_t)(kk0 + kk) * DIMC;
            Bt[t][kk]       = (bf16)wr[t];
            Bt[t + 128][kk] = (bf16)wr[t + 128];
        }
        if (ks < 7)
            __builtin_prefetch(ctx + (size_t)(row0 + (t & 63)) * DIMC + kk0 + 32, 0, 1);
#if USE_ASYNC
        async_wait_all();
#endif
        __syncthreads();

        bf16x16 af = load_a_frag(&As[w * 16][0], 48, lane);
#pragma unroll
        for (int nt = 0; nt < 16; ++nt) {
            bf16x16 bfp = load_bt_frag(&Bt[nt * 16][0], 40, lane);
            acc[nt] = __builtin_amdgcn_wmma_f32_16x16x32_bf16(
                false, af, false, bfp, (short)0, acc[nt], false, false);
        }
        __syncthreads();
    }

#pragma unroll
    for (int nt = 0; nt < 16; ++nt) {
#pragma unroll
        for (int r = 0; r < 8; ++r) {
            int gRow = row0 + w * 16 + r + 8 * (lane >> 4);
            int cg = nt * 16 + (lane & 15);
            out[(size_t)gRow * DIMC + cg] = acc[nt][r] + bproj[cg];
        }
    }
}

// ---------------- launch ----------------

extern "C" void kernel_launch(void* const* d_in, const int* in_sizes, int n_in,
                              void* d_out, int out_size, void* d_ws, size_t ws_size,
                              hipStream_t stream)
{
    (void)in_sizes; (void)n_in; (void)out_size; (void)ws_size;
    const float* x          = (const float*)d_in[0];
    const float* mask       = (const float*)d_in[1];
    const float* w_qkv      = (const float*)d_in[2];
    const float* w_proj     = (const float*)d_in[3];
    const float* b_proj     = (const float*)d_in[4];
    const float* bias_table = (const float*)d_in[5];
    const int*   rel_idx    = (const int*)d_in[6];
    float* out = (float*)d_out;

    bf16* qkv_ws = (bf16*)d_ws;                 // q | k | v, each PERQ elements
    bf16* q_ws = qkv_ws;
    bf16* k_ws = q_ws + PERQ;
    bf16* v_ws = k_ws + PERQ;
    bf16* ctx  = v_ws + PERQ;                   // BWIN*NTOK*DIMC == PERQ elements

    qkv_gemm_kernel<<<dim3(MTILES), 256, 0, stream>>>(x, w_qkv, qkv_ws);
    attn_kernel<<<dim3(BWIN * HEADS), 128, 0, stream>>>(q_ws, k_ws, v_ws, mask,
                                                        bias_table, rel_idx, ctx);
    proj_gemm_kernel<<<dim3(MTILES), 128, 0, stream>>>(ctx, w_proj, b_proj, out);
}
